// TinyGridSpecialist_66735201845456
// MI455X (gfx1250) — matrix operand order repaired and checked by reading, not compiled, not run on hardware
//
#include <hip/hip_runtime.h>
#include <hip/hip_bf16.h>
#include <math.h>

typedef __attribute__((ext_vector_type(16))) _Float16 v16h;
typedef __attribute__((ext_vector_type(8)))  float    v8f;

#define WMMA_F16(a,b,c) __builtin_amdgcn_wmma_f32_16x16x32_f16(false,(a),false,(b),(short)0,(c),false,false)

// ---------------------------------------------------------------- utilities

__device__ inline float wave_sum(float v) {
#pragma unroll
  for (int off = 16; off >= 1; off >>= 1) v += __shfl_xor(v, off);
  return v;
}

// A fragment (16x32 f16, MxK), contiguous k order (same as B loaders):
// lane l holds row m=l&15; element e -> k = e + 16*(l>>4)
__device__ inline v16h load_a_frag(const _Float16* __restrict__ a, int lda,
                                   int mbase, int kbase, int kvalid) {
  int lane = threadIdx.x & 31;
  const _Float16* row = a + (mbase + (lane & 15)) * lda + kbase;
  int kh = (lane >> 4) * 16;
  v16h r;
#pragma unroll
  for (int e = 0; e < 16; ++e) {
    int k = kh + e;
    r[e] = (k < kvalid) ? row[k] : (_Float16)0.f;
  }
  return r;
}

// B fragment (32x16 f16, KxN) from N-major storage w[n*ldb + k]
// lane l holds col n=l&15; element e -> k = e + 16*(l>>4)  (contiguous 32B run)
__device__ inline v16h load_b_frag_nm(const _Float16* __restrict__ w, int ldb,
                                      int nbase, int kbase, int kvalid) {
  int lane = threadIdx.x & 31;
  const _Float16* row = w + (nbase + (lane & 15)) * ldb + kbase;
  int k0 = (lane >> 4) * 16;
  v16h r;
#pragma unroll
  for (int e = 0; e < 16; ++e) {
    int k = k0 + e;
    r[e] = (k < kvalid) ? row[k] : (_Float16)0.f;
  }
  return r;
}

// B fragment from K-major storage w[k*ldb + n] (e.g. V matrix [token][dh])
__device__ inline v16h load_b_frag_km(const _Float16* __restrict__ w, int ldb,
                                      int nbase, int kbase, int kvalid) {
  int lane = threadIdx.x & 31;
  int n = nbase + (lane & 15);
  int k0 = (lane >> 4) * 16;
  v16h r;
#pragma unroll
  for (int e = 0; e < 16; ++e) {
    int k = k0 + e;
    r[e] = (k < kvalid) ? w[(kbase + k) * ldb + n] : (_Float16)0.f;
  }
  return r;
}

// Generic M=32 x N x K GEMM: A f16 in LDS (lda=128), B f16 N-major in global,
// f32 out to LDS with optional bias / residual(x, stride 64) / relu.
template<int N, int K, bool RELU, bool RESID>
__device__ inline void gemm_nm(const _Float16* __restrict__ A,
                               const _Float16* __restrict__ Bw,
                               const float* __restrict__ bias,
                               const float* __restrict__ resid,
                               float* __restrict__ out, int ldo) {
  const int TN = N / 16;
  int wid = threadIdx.x >> 5, lane = threadIdx.x & 31;
  for (int t = wid; t < 2 * TN; t += 8) {
    int mt = t / TN, nt = t % TN;
    v8f acc = {};
#pragma unroll
    for (int kk = 0; kk < K; kk += 32) {
      v16h a = load_a_frag(A, 128, mt * 16, kk, 32);
      v16h b = load_b_frag_nm(Bw, K, nt * 16, kk, 32);
      acc = WMMA_F16(a, b, acc);
    }
    int n  = nt * 16 + (lane & 15);
    int m0 = mt * 16 + ((lane >> 4) * 8);
    float bn = bias[n];
#pragma unroll
    for (int r = 0; r < 8; ++r) {
      float v = acc[r] + bn;
      if constexpr (RESID) v += resid[(m0 + r) * 64 + n];
      if constexpr (RELU)  v = fmaxf(v, 0.f);
      out[(m0 + r) * ldo + n] = v;
    }
  }
}

// ---------------------------------------------------------------- prep kernels

__global__ void tgs_cvt_f16(const float* __restrict__ src, _Float16* __restrict__ dst, int n) {
  for (int i = blockIdx.x * blockDim.x + threadIdx.x; i < n; i += gridDim.x * blockDim.x)
    dst[i] = (_Float16)src[i];
}

// hW2 [10,64] -> f16 [16,64], rows 10..15 zero
__global__ void tgs_pad_hw2(const float* __restrict__ hW2, _Float16* __restrict__ dst) {
  for (int i = threadIdx.x; i < 16 * 64; i += blockDim.x) {
    int r = i >> 6, c = i & 63;
    dst[i] = (r < 10) ? (_Float16)hW2[r * 64 + c] : (_Float16)0.f;
  }
}

// row-normalize pattern_memory [100,1600] into f16
__global__ void tgs_prep_patterns(const float* __restrict__ pm, _Float16* __restrict__ pnh) {
  __shared__ float red[8];
  __shared__ float denom;
  int p = blockIdx.x, t = threadIdx.x;
  float s = 0.f;
  for (int i = t; i < 1600; i += 256) { float v = pm[p * 1600 + i]; s += v * v; }
  s = wave_sum(s);
  if ((t & 31) == 0) red[t >> 5] = s;
  __syncthreads();
  if (t == 0) {
    float tot = 0.f;
    for (int i = 0; i < 8; ++i) tot += red[i];
    denom = fmaxf(sqrtf(tot), 1e-8f);
  }
  __syncthreads();
  float d = denom;
  for (int i = t; i < 1600; i += 256)
    pnh[p * 1600 + i] = (_Float16)(pm[p * 1600 + i] / d);
}

// ---------------------------------------------------------------- fused megakernel

__global__ __launch_bounds__(256) void tgs_fused(
    const float* __restrict__ grid,  const float* __restrict__ pos,
    const float* __restrict__ cemb,
    const _Float16* __restrict__ Wqkvh, const float* __restrict__ bqkv,
    const _Float16* __restrict__ Woh,   const float* __restrict__ bo,
    const float* __restrict__ ln1g, const float* __restrict__ ln1b,
    const _Float16* __restrict__ W1h,   const float* __restrict__ b1,
    const _Float16* __restrict__ W2h,   const float* __restrict__ b2,
    const float* __restrict__ ln2g, const float* __restrict__ ln2b,
    const float* __restrict__ pm,   const _Float16* __restrict__ pnh,
    const _Float16* __restrict__ hW1h,  const float* __restrict__ hb1,
    const _Float16* __restrict__ hW2h,  const float* __restrict__ hb2,
    float* __restrict__ out) {
  __shared__ float    lds_x[32][64];     // persistent activations, f32
  __shared__ _Float16 lds_a[32][128];    // generic A-operand staging, f16
  __shared__ _Float16 lds_q[32][192];    // qkv, f16
  __shared__ float    lds_s[4096];       // generic f32 scratch (scores / ffn / head)
  __shared__ int      lds_idx[32];
  __shared__ float    lds_sc[112];
  __shared__ float    red[8];
  __shared__ float    sh_denom;
  __shared__ float    topw[5];
  __shared__ int      topi[5];

  const int b = blockIdx.x;
  const int tid = threadIdx.x, wid = tid >> 5, lane = tid & 31;
  const float* xf = &lds_x[0][0];
  const _Float16* af = &lds_a[0][0];

  // ---- embed: argmax over 10 colors, x = color_emb[idx] + pos ----
  if (tid < 25) {
    const float* g = grid + (size_t)b * 250 + tid;
    float best = g[0]; int bi = 0;
    for (int c = 1; c < 10; ++c) { float v = g[c * 25]; if (v > best) { best = v; bi = c; } }
    lds_idx[tid] = bi;
  }
  __syncthreads();
  for (int i = tid; i < 2048; i += 256) {
    int s = i >> 6, c = i & 63;
    float v = (s < 25) ? (cemb[lds_idx[s] * 64 + c] + pos[s * 64 + c]) : 0.f;
    lds_x[s][c] = v;
    lds_a[s][c] = (_Float16)v;
  }
  __syncthreads();

  // ---- 2 encoder layers ----
  for (int l = 0; l < 2; ++l) {
    // QKV: [32,64] @ Wqkv^T -> f16 [32,192]
    for (int t = wid; t < 24; t += 8) {
      int mt = t / 12, nt = t % 12;
      v8f acc = {};
#pragma unroll
      for (int kk = 0; kk < 64; kk += 32) {
        v16h a = load_a_frag(af, 128, mt * 16, kk, 32);
        v16h bb = load_b_frag_nm(Wqkvh + l * 12288, 64, nt * 16, kk, 32);
        acc = WMMA_F16(a, bb, acc);
      }
      int n = nt * 16 + (lane & 15);
      int m0 = mt * 16 + ((lane >> 4) * 8);
      float bn = bqkv[l * 192 + n];
#pragma unroll
      for (int r = 0; r < 8; ++r) lds_q[m0 + r][n] = (_Float16)(acc[r] + bn);
    }
    __syncthreads();

    // attention scores per head: Q[32,16] @ K^T -> [4][32][32] f32 (scaled)
    for (int t = wid; t < 16; t += 8) {
      int h = t >> 2, mt = (t >> 1) & 1, nt = t & 1;
      v16h a  = load_a_frag(&lds_q[0][h * 16], 192, mt * 16, 0, 16);
      v16h bb = load_b_frag_nm(&lds_q[0][64 + h * 16], 192, nt * 16, 0, 16);
      v8f acc = {};
      acc = WMMA_F16(a, bb, acc);
      int n = nt * 16 + (lane & 15);
      int m0 = mt * 16 + ((lane >> 4) * 8);
#pragma unroll
      for (int r = 0; r < 8; ++r) lds_s[h * 1024 + (m0 + r) * 32 + n] = acc[r] * 0.25f;
    }
    __syncthreads();

    // softmax over 25 keys -> att f16 into lds_a[m][h*32+n] (padding zeroed)
    if (tid < 128) {
      int h = tid >> 5, m = tid & 31;
      if (m < 25) {
        const float* row = lds_s + h * 1024 + m * 32;
        float mx = row[0];
        for (int n = 1; n < 25; ++n) mx = fmaxf(mx, row[n]);
        float sum = 0.f;
        float e[25];
        for (int n = 0; n < 25; ++n) { e[n] = __expf(row[n] - mx); sum += e[n]; }
        float inv = 1.f / sum;
        for (int n = 0; n < 25; ++n) lds_a[m][h * 32 + n] = (_Float16)(e[n] * inv);
        for (int n = 25; n < 32; ++n) lds_a[m][h * 32 + n] = (_Float16)0.f;
      } else {
        for (int n = 0; n < 32; ++n) lds_a[m][h * 32 + n] = (_Float16)0.f;
      }
    }
    __syncthreads();

    // att @ V -> o f32 [32][64] in lds_s
    {
      int t = wid;  // 8 tiles, one per wave
      int h = t >> 1, mt = t & 1;
      v16h a  = load_a_frag(&lds_a[0][h * 32], 128, mt * 16, 0, 32);
      v16h bb = load_b_frag_km(&lds_q[0][128 + h * 16], 192, 0, 0, 32);
      v8f acc = {};
      acc = WMMA_F16(a, bb, acc);
      int n = lane & 15;
      int m0 = mt * 16 + ((lane >> 4) * 8);
#pragma unroll
      for (int r = 0; r < 8; ++r) lds_s[(m0 + r) * 64 + h * 16 + n] = acc[r];
    }
    __syncthreads();

    // stage o -> lds_a f16
    for (int i = tid; i < 2048; i += 256) lds_a[i >> 6][i & 63] = (_Float16)lds_s[i];
    __syncthreads();

    // Wo projection + residual -> lds_s f32
    gemm_nm<64, 64, false, true>(af, Woh + l * 4096, bo + l * 64, xf, lds_s, 64);
    __syncthreads();

    // LN1 -> lds_x / lds_a
    if (tid < 25) {
      const float* row = lds_s + tid * 64;
      float m = 0.f;
      for (int c = 0; c < 64; ++c) m += row[c];
      m *= (1.f / 64.f);
      float v = 0.f;
      for (int c = 0; c < 64; ++c) { float d = row[c] - m; v += d * d; }
      float rs = rsqrtf(v * (1.f / 64.f) + 1e-5f);
      for (int c = 0; c < 64; ++c) {
        float o = (row[c] - m) * rs * ln1g[l * 64 + c] + ln1b[l * 64 + c];
        lds_x[tid][c] = o;
        lds_a[tid][c] = (_Float16)o;
      }
    }
    __syncthreads();

    // FFN1: relu(x @ W1^T + b1) -> lds_s [32][128]
    gemm_nm<128, 64, true, false>(af, W1h + l * 8192, b1 + l * 128, xf, lds_s, 128);
    __syncthreads();
    for (int i = tid; i < 4096; i += 256) lds_a[i >> 7][i & 127] = (_Float16)lds_s[i];
    __syncthreads();

    // FFN2 + residual -> lds_s [32][64]
    gemm_nm<64, 128, false, true>(af, W2h + l * 8192, b2 + l * 64, xf, lds_s, 64);
    __syncthreads();

    // LN2 -> lds_x / lds_a
    if (tid < 25) {
      const float* row = lds_s + tid * 64;
      float m = 0.f;
      for (int c = 0; c < 64; ++c) m += row[c];
      m *= (1.f / 64.f);
      float v = 0.f;
      for (int c = 0; c < 64; ++c) { float d = row[c] - m; v += d * d; }
      float rs = rsqrtf(v * (1.f / 64.f) + 1e-5f);
      for (int c = 0; c < 64; ++c) {
        float o = (row[c] - m) * rs * ln2g[l * 64 + c] + ln2b[l * 64 + c];
        lds_x[tid][c] = o;
        lds_a[tid][c] = (_Float16)o;
      }
    }
    __syncthreads();
  }

  // ---- ||x|| ----
  {
    float s = 0.f;
    for (int i = tid; i < 1600; i += 256) { float v = xf[i]; s += v * v; }
    s = wave_sum(s);
    if ((tid & 31) == 0) red[tid >> 5] = s;
    __syncthreads();
    if (tid == 0) {
      float tot = 0.f;
      for (int i = 0; i < 8; ++i) tot += red[i];
      sh_denom = fmaxf(sqrtf(tot), 1e-8f);
    }
    __syncthreads();
  }

  // ---- cosine scores vs 100 normalized patterns ----
  for (int p = wid; p < 100; p += 8) {
    float dot = 0.f;
    const _Float16* pr = pnh + p * 1600;
    for (int i = lane; i < 1600; i += 32) dot += xf[i] * (float)pr[i];
    dot = wave_sum(dot);
    if (lane == 0) lds_sc[p] = dot / sh_denom;
  }
  __syncthreads();

  // ---- top-5 + softmax weights ----
  if (tid == 0) {
    float tv[5]; int ti[5];
    for (int j = 0; j < 5; ++j) {
      float best = -3.4e38f; int bi = 0;
      for (int p = 0; p < 100; ++p) { float v = lds_sc[p]; if (v > best) { best = v; bi = p; } }
      tv[j] = best; ti[j] = bi;
      lds_sc[bi] = -3.4e38f;
    }
    float sum = 0.f, e[5];
    for (int j = 0; j < 5; ++j) { e[j] = __expf(tv[j] - tv[0]); sum += e[j]; }
    for (int j = 0; j < 5; ++j) { topw[j] = e[j] / sum; topi[j] = ti[j]; }
  }
  __syncthreads();

  // ---- x += 0.5 * sum_j w_j * pattern_memory[top_j] ----
  for (int i = tid; i < 1600; i += 256) {
    float acc = 0.f;
#pragma unroll
    for (int j = 0; j < 5; ++j) acc += topw[j] * pm[(size_t)topi[j] * 1600 + i];
    int s = i >> 6, c = i & 63;
    float nx = lds_x[s][c] + 0.5f * acc;
    lds_x[s][c] = nx;
    lds_a[s][c] = (_Float16)nx;
  }
  __syncthreads();

  // ---- head MLP: relu(x @ hW1^T + hb1) ----
  gemm_nm<64, 64, true, false>(af, hW1h, hb1, xf, lds_s, 64);
  __syncthreads();
  for (int i = tid; i < 2048; i += 256) lds_a[i >> 6][i & 63] = (_Float16)lds_s[i];
  __syncthreads();

  // ---- head2: h @ hW2p^T (N padded to 16) -> direct transposed global store ----
  if (wid < 2) {
    int mt = wid;
    v8f acc = {};
#pragma unroll
    for (int kk = 0; kk < 64; kk += 32) {
      v16h a = load_a_frag(af, 128, mt * 16, kk, 32);
      v16h bb = load_b_frag_nm(hW2h, 64, 0, kk, 32);
      acc = WMMA_F16(a, bb, acc);
    }
    int c = lane & 15;
    int m0 = mt * 16 + ((lane >> 4) * 8);
    if (c < 10) {
      float bn = hb2[c];
      float* ob = out + (size_t)b * 250 + c * 25;
#pragma unroll
      for (int r = 0; r < 8; ++r) {
        int s = m0 + r;
        if (s < 25) ob[s] = acc[r] + bn;
      }
    }
  }
}

// ---------------------------------------------------------------- launch

extern "C" void kernel_launch(void* const* d_in, const int* in_sizes, int n_in,
                              void* d_out, int out_size, void* d_ws, size_t ws_size,
                              hipStream_t stream) {
  const float* grid = (const float*)d_in[0];
  const float* pos  = (const float*)d_in[1];
  const float* cemb = (const float*)d_in[2];
  const float* Wqkv = (const float*)d_in[3];
  const float* bqkv = (const float*)d_in[4];
  const float* Wo   = (const float*)d_in[5];
  const float* bo   = (const float*)d_in[6];
  const float* ln1g = (const float*)d_in[7];
  const float* ln1b = (const float*)d_in[8];
  const float* W1   = (const float*)d_in[9];
  const float* b1   = (const float*)d_in[10];
  const float* W2   = (const float*)d_in[11];
  const float* b2   = (const float*)d_in[12];
  const float* ln2g = (const float*)d_in[13];
  const float* ln2b = (const float*)d_in[14];
  const float* pm   = (const float*)d_in[15];
  const float* hW1  = (const float*)d_in[16];
  const float* hb1  = (const float*)d_in[17];
  const float* hW2  = (const float*)d_in[18];
  const float* hb2  = (const float*)d_in[19];
  float* outp = (float*)d_out;

  const int B = in_sizes[0] / 250;

  _Float16* wsh = (_Float16*)d_ws;
  size_t off = 0;
  auto take = [&](size_t n) { _Float16* p = wsh + off; off += (n + 127) & ~(size_t)127; return p; };
  _Float16* Wqkvh = take(2 * 192 * 64);
  _Float16* Woh   = take(2 * 64 * 64);
  _Float16* W1h   = take(2 * 128 * 64);
  _Float16* W2h   = take(2 * 64 * 128);
  _Float16* hW1h  = take(64 * 64);
  _Float16* hW2h  = take(16 * 64);
  _Float16* pnh   = take(100 * 1600);

  tgs_cvt_f16<<<96, 256, 0, stream>>>(Wqkv, Wqkvh, 2 * 192 * 64);
  tgs_cvt_f16<<<32, 256, 0, stream>>>(Wo,   Woh,   2 * 64 * 64);
  tgs_cvt_f16<<<64, 256, 0, stream>>>(W1,   W1h,   2 * 128 * 64);
  tgs_cvt_f16<<<64, 256, 0, stream>>>(W2,   W2h,   2 * 64 * 128);
  tgs_cvt_f16<<<16, 256, 0, stream>>>(hW1,  hW1h,  64 * 64);
  tgs_pad_hw2<<<1, 256, 0, stream>>>(hW2, hW2h);
  tgs_prep_patterns<<<100, 256, 0, stream>>>(pm, pnh);

  tgs_fused<<<B, 256, 0, stream>>>(grid, pos, cemb,
                                   Wqkvh, bqkv, Woh, bo, ln1g, ln1b,
                                   W1h, b1, W2h, b2, ln2g, ln2b,
                                   pm, pnh, hW1h, hb1, hW2h, hb2, outp);
}
